// Block_67327907332707
// MI455X (gfx1250) — matrix operand order repaired and checked by reading, not compiled, not run on hardware
//
#include <hip/hip_runtime.h>
#include <hip/hip_bf16.h>

typedef __bf16 bf16x16 __attribute__((ext_vector_type(16)));
typedef __bf16 bf16x8  __attribute__((ext_vector_type(8)));
typedef float  f32x8   __attribute__((ext_vector_type(8)));

#define DIM   768
#define HEADS 12
#define HD    64
#define HID3  3072
#define QKVD  2304
#define BATCH 8
#define Tn    8
#define HWn   196
#define NTOK  1569
#define MT    12544   // BATCH*HW*T   (temporal tokens)
#define MS    12608   // BATCH*T*197  (spatial tokens)
#define MA    12552   // BATCH*NTOK   (all tokens)
#define NSEQ  197

__device__ __forceinline__ unsigned short f2bf(float f) {
    unsigned int u = __float_as_uint(f);
    u += 0x7FFFu + ((u >> 16) & 1u);
    return (unsigned short)(u >> 16);
}

__device__ __forceinline__ float wave_sum(float v) {
    #pragma unroll
    for (int o = 16; o; o >>= 1) v += __shfl_xor(v, o, 32);
    return v;
}
__device__ __forceinline__ float wave_max(float v) {
    #pragma unroll
    for (int o = 16; o; o >>= 1) v = fmaxf(v, __shfl_xor(v, o, 32));
    return v;
}

// ---------------- fp32 -> bf16 conversion ----------------
__global__ __launch_bounds__(256) void cvt_bf16_kernel(const float* __restrict__ in,
                                                       unsigned short* __restrict__ out, int n) {
    int i = blockIdx.x * 256 + threadIdx.x;
    if (i < n) out[i] = f2bf(in[i]);
}

// ---------------- LayerNorm (one wave per 768-wide row), bf16 out ----------------
// mode 0: temporal input  (row r of xt0 -> x row r + r/1568 + 1)
// mode 1: xs gather       (row m: p==0 -> cls of x; else xt row b*1568 + (p-1)*8 + t)
// mode 2: plain rows of X
__global__ __launch_bounds__(256) void ln_kernel(const float* __restrict__ X,
                                                 const float* __restrict__ XT,
                                                 const float* __restrict__ g,
                                                 const float* __restrict__ bb,
                                                 unsigned short* __restrict__ out,
                                                 int mode, int nrows) {
    int wv = threadIdx.x >> 5, lane = threadIdx.x & 31;
    int row = blockIdx.x * 8 + wv;
    if (row >= nrows) return;
    const float* src;
    if (mode == 0) {
        int b = row / 1568;
        src = X + (size_t)(row + b + 1) * DIM;
    } else if (mode == 1) {
        int s = row / NSEQ, p = row % NSEQ;
        int b = s >> 3, t = s & 7;
        if (p == 0) src = X + (size_t)b * NTOK * DIM;
        else        src = XT + (size_t)(b * (HWn * Tn) + (p - 1) * Tn + t) * DIM;
    } else {
        src = X + (size_t)row * DIM;
    }
    float v[24];
    float s1 = 0.f;
    #pragma unroll
    for (int i = 0; i < 24; i++) { v[i] = src[lane + i * 32]; s1 += v[i]; }
    s1 = wave_sum(s1);
    float mean = s1 * (1.f / 768.f);
    float s2 = 0.f;
    #pragma unroll
    for (int i = 0; i < 24; i++) { float d = v[i] - mean; s2 += d * d; }
    s2 = wave_sum(s2);
    float rstd = rsqrtf(s2 * (1.f / 768.f) + 1e-5f);
    unsigned short* o = out + (size_t)row * DIM;
    #pragma unroll
    for (int i = 0; i < 24; i++) {
        int c = lane + i * 32;
        o[c] = f2bf((v[i] - mean) * rstd * g[c] + bb[c]);
    }
}

// ---------------- WMMA GEMM: C[M,N] = A[M,K] * W[N,K]^T (+bias)(GELU)(+res) ----------------
// block tile 128x128, 8 waves, wave tile 32x64 (2x4 fragments of 16x16), K step 32
__global__ __launch_bounds__(256) void gemm_bf16_wmma(const unsigned short* __restrict__ A,
                                                      const unsigned short* __restrict__ Bw,
                                                      const float* __restrict__ bias,
                                                      float* __restrict__ Cf,
                                                      unsigned short* __restrict__ Cb,
                                                      const float* __restrict__ Res,
                                                      int M, int N, int K, int gelu) {
    const int n0 = blockIdx.x * 128;
    const int m0 = blockIdx.y * 128;
    const int wv = threadIdx.x >> 5;
    const int lane = threadIdx.x & 31;
    const int kh = lane >> 4;     // K-half select
    const int l  = lane & 15;     // M (A) / N (B,C) index
    const int waveM = wv & 3;
    const int waveN = wv >> 2;
    const int mbase = m0 + waveM * 32;
    const int nbase = n0 + waveN * 64;

    f32x8 acc[2][4];
    #pragma unroll
    for (int i = 0; i < 2; i++)
        #pragma unroll
        for (int j = 0; j < 4; j++)
            #pragma unroll
            for (int e = 0; e < 8; e++) acc[i][j][e] = 0.f;

    int arow[2];
    #pragma unroll
    for (int mf = 0; mf < 2; mf++) {
        int r = mbase + mf * 16 + l;
        arow[mf] = (r < M) ? r : (M - 1);   // clamp tail rows (stores are guarded)
    }

    for (int k0 = 0; k0 < K; k0 += 32) {
        bf16x16 af[2];
        #pragma unroll
        for (int mf = 0; mf < 2; mf++) {
            // lane<16: K = k0+[0..7] and k0+16+[0..7]; lane>=16: +8 (ISA 16-bit A 16x32 layout)
            const unsigned short* p = A + (size_t)arow[mf] * K + k0 + kh * 8;
            bf16x8 c0 = *reinterpret_cast<const bf16x8*>(p);
            bf16x8 c1 = *reinterpret_cast<const bf16x8*>(p + 16);
            af[mf] = __builtin_shufflevector(c0, c1, 0,1,2,3,4,5,6,7,8,9,10,11,12,13,14,15);
        }
        #pragma unroll
        for (int nf = 0; nf < 4; nf++) {
            // B[k][n] = W[n][k]: lane<16 holds K=k0+[0..7] and k0+16+[0..7] of row n; lane>=16: +8
            const unsigned short* p = Bw + (size_t)(nbase + nf * 16 + l) * K + k0 + kh * 8;
            bf16x8 c0 = *reinterpret_cast<const bf16x8*>(p);
            bf16x8 c1 = *reinterpret_cast<const bf16x8*>(p + 16);
            bf16x16 bfr = __builtin_shufflevector(c0, c1, 0,1,2,3,4,5,6,7,8,9,10,11,12,13,14,15);
            #pragma unroll
            for (int mf = 0; mf < 2; mf++)
                acc[mf][nf] = __builtin_amdgcn_wmma_f32_16x16x32_bf16(
                    false, af[mf], false, bfr, (short)0, acc[mf][nf], false, false);
        }
    }

    #pragma unroll
    for (int mf = 0; mf < 2; mf++) {
        #pragma unroll
        for (int j = 0; j < 8; j++) {
            int row = mbase + mf * 16 + kh * 8 + j;   // C layout: VGPR j -> M=j (+8 for hi lanes)
            if (row < M) {
                #pragma unroll
                for (int nf = 0; nf < 4; nf++) {
                    int col = nbase + nf * 16 + l;
                    float v = acc[mf][nf][j];
                    if (bias) v += bias[col];
                    if (gelu) v = 0.5f * v * (1.f + erff(v * 0.70710678118654752f));
                    if (Res)  v += Res[(size_t)row * N + col];
                    if (Cf)   Cf[(size_t)row * N + col] = v;
                    if (Cb)   Cb[(size_t)row * N + col] = f2bf(v);
                }
            }
        }
    }
}

// ---------------- temporal attention: 1568 blocks, thread = (head, t), T=8, d=64 ----------------
__global__ __launch_bounds__(96) void attn_temporal_kernel(const float* __restrict__ qkv,
                                                           unsigned short* __restrict__ outb) {
    int bb = blockIdx.x;
    int tid = threadIdx.x;
    int hh = tid >> 3, t = tid & 7;
    const float* qr = qkv + (size_t)(bb * 8 + t) * QKVD + hh * HD;
    float q[HD];
    #pragma unroll
    for (int c = 0; c < HD; c++) q[c] = qr[c];
    float s[8];
    #pragma unroll
    for (int u = 0; u < 8; u++) {
        const float* kr = qkv + (size_t)(bb * 8 + u) * QKVD + DIM + hh * HD;
        float d = 0.f;
        #pragma unroll
        for (int c = 0; c < HD; c++) d += q[c] * kr[c];
        s[u] = d * 0.125f;
    }
    float mx = s[0];
    #pragma unroll
    for (int u = 1; u < 8; u++) mx = fmaxf(mx, s[u]);
    float sum = 0.f;
    #pragma unroll
    for (int u = 0; u < 8; u++) { s[u] = expf(s[u] - mx); sum += s[u]; }
    float inv = 1.f / sum;
    float o[HD];
    #pragma unroll
    for (int c = 0; c < HD; c++) o[c] = 0.f;
    #pragma unroll
    for (int u = 0; u < 8; u++) {
        const float* vr = qkv + (size_t)(bb * 8 + u) * QKVD + 2 * DIM + hh * HD;
        float a = s[u] * inv;
        #pragma unroll
        for (int c = 0; c < HD; c++) o[c] += a * vr[c];
    }
    unsigned short* ob = outb + (size_t)(bb * 8 + t) * DIM + hh * HD;
    #pragma unroll
    for (int c = 0; c < HD; c++) ob[c] = f2bf(o[c]);
}

// ---------------- spatial attention: grid (64,12), block 256, K/V in LDS ----------------
__global__ __launch_bounds__(256) void attn_spatial_kernel(const float* __restrict__ qkv,
                                                           unsigned short* __restrict__ outb) {
    extern __shared__ float sm[];
    float* kk = sm;                       // 197*65
    float* vv = sm + NSEQ * 65;           // 197*65
    float* sc = vv + NSEQ * 65;           // 256
    __shared__ float red[8];
    int s = blockIdx.x, hh = blockIdx.y;
    size_t base = (size_t)s * NSEQ;
    int tid = threadIdx.x, lane = tid & 31, wv = tid >> 5;

    for (int idx = tid; idx < NSEQ * HD; idx += 256) {
        int r = idx >> 6, c = idx & 63;
        size_t ro = (base + r) * QKVD + hh * HD + c;
        kk[r * 65 + c] = qkv[ro + DIM];
        vv[r * 65 + c] = qkv[ro + 2 * DIM];
    }
    __syncthreads();

    for (int i = 0; i < NSEQ; i++) {
        const float* qr = qkv + (base + i) * QKVD + hh * HD;
        float sv = -1e30f;
        if (tid < NSEQ) {
            float d = 0.f;
            #pragma unroll 8
            for (int c = 0; c < HD; c++) d += qr[c] * kk[tid * 65 + c];
            sv = d * 0.125f;
        }
        float mx = wave_max(sv);
        if (lane == 0) red[wv] = mx;
        __syncthreads();
        mx = red[0];
        #pragma unroll
        for (int w = 1; w < 8; w++) mx = fmaxf(mx, red[w]);
        float ex = (tid < NSEQ) ? expf(sv - mx) : 0.f;
        sc[tid] = ex;
        float ps = wave_sum(ex);
        __syncthreads();
        if (lane == 0) red[wv] = ps;
        __syncthreads();
        float sum = red[0];
        #pragma unroll
        for (int w = 1; w < 8; w++) sum += red[w];
        float inv = 1.f / sum;
        if (tid < HD) {
            float acc = 0.f;
            for (int j = 0; j < NSEQ; j++) acc += sc[j] * vv[j * 65 + tid];
            outb[(base + i) * DIM + hh * HD + tid] = f2bf(acc * inv);
        }
        __syncthreads();
    }
}

// ---------------- glue kernels ----------------
__global__ __launch_bounds__(256) void add_xt_kernel(const float* __restrict__ X,
                                                     const float* __restrict__ rt,
                                                     float* __restrict__ xt, int n) {
    int idx = blockIdx.x * 256 + threadIdx.x;
    if (idx >= n) return;
    int r = idx / DIM, c = idx % DIM;
    int b = r / 1568;
    xt[idx] = X[(size_t)(r + b + 1) * DIM + c] + rt[idx];
}

__global__ __launch_bounds__(256) void cls_mean_kernel(const float* __restrict__ rs,
                                                       float* __restrict__ cls, int n) {
    int idx = blockIdx.x * 256 + threadIdx.x;
    if (idx >= n) return;
    int b = idx / DIM, c = idx % DIM;
    float a = 0.f;
    #pragma unroll
    for (int t = 0; t < 8; t++) a += rs[(size_t)((b * 8 + t) * NSEQ) * DIM + c];
    cls[idx] = a * 0.125f;
}

__global__ __launch_bounds__(256) void build_xo_kernel(const float* __restrict__ X,
                                                       const float* __restrict__ xt,
                                                       const float* __restrict__ rs,
                                                       const float* __restrict__ cls,
                                                       float* __restrict__ xo, int n) {
    int idx = blockIdx.x * 256 + threadIdx.x;
    if (idx >= n) return;
    int m = idx / DIM, c = idx % DIM;
    int b = m / NTOK, p = m % NTOK;
    float v;
    if (p == 0) {
        v = X[idx] + cls[b * DIM + c];
    } else {
        int j = p - 1, q = j >> 3, t = j & 7;
        v = xt[(size_t)(b * 1568 + j) * DIM + c] +
            rs[(size_t)((b * 8 + t) * NSEQ + 1 + q) * DIM + c];
    }
    xo[idx] = v;
}

// ---------------- launch ----------------
extern "C" void kernel_launch(void* const* d_in, const int* in_sizes, int n_in,
                              void* d_out, int out_size, void* d_ws, size_t ws_size,
                              hipStream_t stream) {
    const float* x       = (const float*)d_in[0];
    const float* ln1_g   = (const float*)d_in[1];
    const float* ln1_b   = (const float*)d_in[2];
    const float* lnt_g   = (const float*)d_in[3];
    const float* lnt_b   = (const float*)d_in[4];
    const float* ln2_g   = (const float*)d_in[5];
    const float* ln2_b   = (const float*)d_in[6];
    const float* qkv_w   = (const float*)d_in[7];
    const float* proj_w  = (const float*)d_in[8];
    const float* proj_b  = (const float*)d_in[9];
    const float* tqkv_w  = (const float*)d_in[10];
    const float* tproj_w = (const float*)d_in[11];
    const float* tproj_b = (const float*)d_in[12];
    const float* tfc_w   = (const float*)d_in[13];
    const float* tfc_b   = (const float*)d_in[14];
    const float* fc1_w   = (const float*)d_in[15];
    const float* fc1_b   = (const float*)d_in[16];
    const float* fc2_w   = (const float*)d_in[17];
    const float* fc2_b   = (const float*)d_in[18];
    float* out = (float*)d_out;

    char* ws = (char*)d_ws;
    size_t off = 0;
    auto take = [&](size_t bytes) -> char* {
        char* p = ws + off;
        off = (off + bytes + 255) & ~(size_t)255;
        return p;
    };
    unsigned short* wb_tqkv = (unsigned short*)take((size_t)QKVD * DIM * 2);
    unsigned short* wb_tprj = (unsigned short*)take((size_t)DIM * DIM * 2);
    unsigned short* wb_tfc  = (unsigned short*)take((size_t)DIM * DIM * 2);
    unsigned short* wb_qkv  = (unsigned short*)take((size_t)QKVD * DIM * 2);
    unsigned short* wb_prj  = (unsigned short*)take((size_t)DIM * DIM * 2);
    unsigned short* wb_fc1  = (unsigned short*)take((size_t)HID3 * DIM * 2);
    unsigned short* wb_fc2  = (unsigned short*)take((size_t)DIM * HID3 * 2);
    unsigned short* BF1 = (unsigned short*)take((size_t)MS * DIM * 2);   // ln_t / rt1 / xs_ln / ln2
    unsigned short* BF2 = (unsigned short*)take((size_t)MS * DIM * 2);   // attn_t / attn_s
    unsigned short* BF3 = (unsigned short*)take((size_t)MA * HID3 * 2);  // gelu(fc1) bf16
    float* F32A = (float*)take((size_t)MS * QKVD * 4);                   // qkv_t / rt / qkv_s / rs+xo
    float* XT   = (float*)take((size_t)MT * DIM * 4);
    float* CLS  = (float*)take((size_t)BATCH * DIM * 4);
    float* RS   = F32A;
    float* XO   = F32A + 9683200;   // past rs (9,682,944 floats), 256B aligned

    const int nW1 = QKVD * DIM, nW2 = DIM * DIM, nW3 = HID3 * DIM;
    cvt_bf16_kernel<<<(nW1 + 255) / 256, 256, 0, stream>>>(tqkv_w,  wb_tqkv, nW1);
    cvt_bf16_kernel<<<(nW2 + 255) / 256, 256, 0, stream>>>(tproj_w, wb_tprj, nW2);
    cvt_bf16_kernel<<<(nW2 + 255) / 256, 256, 0, stream>>>(tfc_w,   wb_tfc,  nW2);
    cvt_bf16_kernel<<<(nW1 + 255) / 256, 256, 0, stream>>>(qkv_w,   wb_qkv,  nW1);
    cvt_bf16_kernel<<<(nW2 + 255) / 256, 256, 0, stream>>>(proj_w,  wb_prj,  nW2);
    cvt_bf16_kernel<<<(nW3 + 255) / 256, 256, 0, stream>>>(fc1_w,   wb_fc1,  nW3);
    cvt_bf16_kernel<<<(nW3 + 255) / 256, 256, 0, stream>>>(fc2_w,   wb_fc2,  nW3);

    // --- temporal branch ---
    ln_kernel<<<(MT + 7) / 8, 256, 0, stream>>>(x, nullptr, lnt_g, lnt_b, BF1, 0, MT);
    gemm_bf16_wmma<<<dim3(QKVD / 128, (MT + 127) / 128), 256, 0, stream>>>(
        BF1, wb_tqkv, nullptr, F32A, nullptr, nullptr, MT, QKVD, DIM, 0);
    attn_temporal_kernel<<<1568, 96, 0, stream>>>(F32A, BF2);
    gemm_bf16_wmma<<<dim3(DIM / 128, (MT + 127) / 128), 256, 0, stream>>>(
        BF2, wb_tprj, tproj_b, nullptr, BF1, nullptr, MT, DIM, DIM, 0);
    gemm_bf16_wmma<<<dim3(DIM / 128, (MT + 127) / 128), 256, 0, stream>>>(
        BF1, wb_tfc, tfc_b, F32A, nullptr, nullptr, MT, DIM, DIM, 0);
    {
        int n = MT * DIM;
        add_xt_kernel<<<(n + 255) / 256, 256, 0, stream>>>(x, F32A, XT, n);
    }

    // --- spatial branch ---
    ln_kernel<<<(MS + 7) / 8, 256, 0, stream>>>(x, XT, ln1_g, ln1_b, BF1, 1, MS);
    gemm_bf16_wmma<<<dim3(QKVD / 128, (MS + 127) / 128), 256, 0, stream>>>(
        BF1, wb_qkv, nullptr, F32A, nullptr, nullptr, MS, QKVD, DIM, 0);
    {
        size_t shbytes = (size_t)(NSEQ * 65 * 2 + 256) * 4;   // 103,464 B
        hipFuncSetAttribute((const void*)attn_spatial_kernel,
                            hipFuncAttributeMaxDynamicSharedMemorySize, (int)shbytes);
        attn_spatial_kernel<<<dim3(64, 12), 256, shbytes, stream>>>(F32A, BF2);
    }
    gemm_bf16_wmma<<<dim3(DIM / 128, (MS + 127) / 128), 256, 0, stream>>>(
        BF2, wb_prj, proj_b, RS, nullptr, nullptr, MS, DIM, DIM, 0);
    {
        int n = BATCH * DIM;
        cls_mean_kernel<<<(n + 255) / 256, 256, 0, stream>>>(RS, CLS, n);
        int n2 = MA * DIM;
        build_xo_kernel<<<(n2 + 255) / 256, 256, 0, stream>>>(x, XT, RS, CLS, XO, n2);
    }

    // --- MLP ---
    ln_kernel<<<(MA + 7) / 8, 256, 0, stream>>>(XO, nullptr, ln2_g, ln2_b, BF1, 2, MA);
    gemm_bf16_wmma<<<dim3(HID3 / 128, (MA + 127) / 128), 256, 0, stream>>>(
        BF1, wb_fc1, fc1_b, nullptr, BF3, nullptr, MA, HID3, DIM, 1);
    gemm_bf16_wmma<<<dim3(DIM / 128, (MA + 127) / 128), 256, 0, stream>>>(
        BF3, wb_fc2, fc2_b, out, nullptr, XO, MA, DIM, HID3, 0);
}
